// MultiHeadAttention_36782099923473
// MI455X (gfx1250) — matrix operand order repaired and checked by reading, not compiled, not run on hardware
//
#include <hip/hip_runtime.h>
#include <hip/hip_fp16.h>

// MI455X / gfx1250, wave32. Fused MHA: f16 WMMA everywhere, flash-style attention.

typedef __attribute__((ext_vector_type(16))) _Float16 v16h;
typedef __attribute__((ext_vector_type(8)))  _Float16 v8h;
typedef __attribute__((ext_vector_type(8)))  float    v8f;
typedef __attribute__((ext_vector_type(4)))  float    f32x4;

constexpr int Bn = 8, Tn = 2048, Dn = 1024, Hn = 16, DKn = 64;
constexpr int HD = Hn * DKn;           // 1024
constexpr float PADV = -4294967295.0f; // -2^32+1, matches reference PAD

__device__ inline v8f wmma16x16x32(v16h a, v16h b, v8f c) {
  return __builtin_amdgcn_wmma_f32_16x16x32_f16(false, a, false, b, (short)0, c,
                                                false, false);
}
__device__ inline v8f zero8() {
  v8f z;
#pragma unroll
  for (int i = 0; i < 8; ++i) z[i] = 0.0f;
  return z;
}

// ---------------------------------------------------------------------------
// Convert + transpose weights to f16: in [batch, rows, cols] f32 ->
// out [batch, cols, rows] f16 (K-dim contiguous for WMMA B-fragments).
// ---------------------------------------------------------------------------
__global__ void cvt_transpose_kernel(const float* __restrict__ in,
                                     _Float16* __restrict__ out,
                                     int batch, int rows, int cols) {
  size_t total = (size_t)batch * rows * cols;
  size_t idx = (size_t)blockIdx.x * blockDim.x + threadIdx.x;
  if (idx >= total) return;
  int c = (int)(idx % cols);
  size_t t = idx / cols;
  int r = (int)(t % rows);
  int bt = (int)(t / rows);
  out[((size_t)bt * cols + c) * rows + r] = (_Float16)in[idx];
}

// ---------------------------------------------------------------------------
// Per-head projection GEMM: x[B,T,D] (f32) @ WT[H,DK,D] (f16, pre-transposed)
// + bias -> f16. MODE 0: Q -> [B,H,T,DK]; MODE 1: K -> [B,H,T,DK] + keymask;
// MODE 2: V -> [B,H,DK,T] (transposed for P*V B-fragments).
// One wave computes a 16(t) x 64(DK) tile; K-loop over D in steps of 32.
// ---------------------------------------------------------------------------
template <int MODE>
__global__ __launch_bounds__(128) void proj_kernel(
    const float* __restrict__ x, const _Float16* __restrict__ WT,
    const float* __restrict__ bias, _Float16* __restrict__ out,
    float* __restrict__ keymask) {
  const int lane = threadIdx.x & 31;
  const int wave = threadIdx.x >> 5;
  const int wid = blockIdx.x * 4 + wave;
  const int ntile = Tn / 16;  // 128
  const int t0 = (wid % ntile) * 16;
  const int h = (wid / ntile) % Hn;
  const int b = wid / (ntile * Hn);
  const int n = lane & 15;
  const int half = lane >> 4;

  const float* arow = x + ((size_t)b * Tn + (t0 + n)) * Dn;  // A row per lane
  const _Float16* wbase = WT + (size_t)h * DKn * Dn;

  v8f acc[4];
#pragma unroll
  for (int nb = 0; nb < 4; ++nb) acc[nb] = zero8();

  for (int kk = 0; kk < Dn; kk += 32) {
    v16h a;
#pragma unroll
    for (int g = 0; g < 2; ++g) {  // A K-pattern: i<8 -> kk+i+8*half ; i>=8 -> +16
      const float* p = arow + kk + g * 16 + half * 8;
      f32x4 f0 = *(const f32x4*)(p);
      f32x4 f1 = *(const f32x4*)(p + 4);
#pragma unroll
      for (int j = 0; j < 4; ++j) {
        a[g * 8 + j] = (_Float16)f0[j];
        a[g * 8 + 4 + j] = (_Float16)f1[j];
      }
    }
#pragma unroll
    for (int nb = 0; nb < 4; ++nb) {
      const _Float16* bp = wbase + (size_t)(nb * 16 + n) * Dn + kk + half * 8;
      v8h blo = *(const v8h*)bp, bhi = *(const v8h*)(bp + 16);
      v16h bf;  // B[k][col], same K-pattern as A
#pragma unroll
      for (int j = 0; j < 8; ++j) { bf[j] = blo[j]; bf[8 + j] = bhi[j]; }
      acc[nb] = wmma16x16x32(a, bf, acc[nb]);
    }
  }

#pragma unroll
  for (int nb = 0; nb < 4; ++nb) {
    float bv = bias[h * DKn + nb * 16 + n];
#pragma unroll
    for (int v = 0; v < 8; ++v) acc[nb][v] += bv;
  }

  if (MODE == 1) {  // key padding mask: sign(sum |ks|) over DK per (b,h,t)
#pragma unroll
    for (int v = 0; v < 8; ++v) {
      float s = 0.0f;
#pragma unroll
      for (int nb = 0; nb < 4; ++nb) s += fabsf(acc[nb][v]);
      s += __shfl_xor(s, 1);
      s += __shfl_xor(s, 2);
      s += __shfl_xor(s, 4);
      s += __shfl_xor(s, 8);
      if (n == 0)
        keymask[((size_t)b * Hn + h) * Tn + t0 + v + half * 8] =
            (s != 0.0f) ? 1.0f : 0.0f;
    }
  }

  if (MODE < 2) {  // [B,H,T,DK]
    _Float16* ob = out + (((size_t)b * Hn + h) * Tn + t0) * DKn;
#pragma unroll
    for (int nb = 0; nb < 4; ++nb)
#pragma unroll
      for (int v = 0; v < 8; ++v)
        ob[(size_t)(v + half * 8) * DKn + nb * 16 + n] = (_Float16)acc[nb][v];
  } else {  // [B,H,DK,T]
    _Float16* ob = out + ((size_t)b * Hn + h) * (size_t)DKn * Tn + t0;
#pragma unroll
    for (int nb = 0; nb < 4; ++nb)
#pragma unroll
      for (int v = 0; v < 8; ++v)
        ob[(size_t)(nb * 16 + n) * Tn + v + half * 8] = (_Float16)acc[nb][v];
  }
}

// ---------------------------------------------------------------------------
// Flash attention: one wave per 16-query tile per (b,h). Streams 64-key
// chunks: S = Q K^T via 8 WMMAs, online softmax in C-layout registers
// (causal compares only on diagonal chunks), P transposed to A-layout
// through a per-wave LDS tile, O += P V via 8 WMMAs.
// ---------------------------------------------------------------------------
__global__ __launch_bounds__(128) void attn_kernel(
    const _Float16* __restrict__ qs,   // [B,H,T,DK]
    const _Float16* __restrict__ ks,   // [B,H,T,DK]
    const _Float16* __restrict__ vsT,  // [B,H,DK,T]
    const float* __restrict__ keymask, // [B,H,T]
    _Float16* __restrict__ heads) {    // [B,T,H*DK]
  // P tile per wave: 16 rows x 64 keys, row stride 72 halves (144B = 9*16B:
  // v8h reads stay 16B aligned; m*36 mod 64 covers 16 distinct banks).
  __shared__ _Float16 lds[4][16 * 72];
  const int lane = threadIdx.x & 31;
  const int wave = threadIdx.x >> 5;
  const int wid = blockIdx.x * 4 + wave;
  const int ntile = Tn / 16;
  const int q0 = (wid % ntile) * 16;
  const int h = (wid / ntile) % Hn;
  const int b = wid / (ntile * Hn);
  const int n = lane & 15;
  const int half = lane >> 4;

  const _Float16* qbase = qs + ((size_t)b * Hn + h) * (size_t)Tn * DKn;
  const _Float16* kbase = ks + ((size_t)b * Hn + h) * (size_t)Tn * DKn;
  const _Float16* vbase = vsT + ((size_t)b * Hn + h) * (size_t)DKn * Tn;
  const float* mbase = keymask + ((size_t)b * Hn + h) * Tn;

  // Q fragments for the two 32-wide feature chunks (DK = 64)
  v16h qa[2];
  const _Float16* qrow = qbase + (size_t)(q0 + n) * DKn;
#pragma unroll
  for (int ec = 0; ec < 2; ++ec) {
    v8h lo = *(const v8h*)(qrow + ec * 32 + half * 8);
    v8h hi = *(const v8h*)(qrow + ec * 32 + 16 + half * 8);
#pragma unroll
    for (int j = 0; j < 8; ++j) {
      qa[ec][j] = lo[j];
      qa[ec][8 + j] = hi[j];
    }
  }

  float mi[8], li[8];
#pragma unroll
  for (int v = 0; v < 8; ++v) {
    mi[v] = -3.0e38f;
    li[v] = 0.0f;
  }
  v8f o[4];
#pragma unroll
  for (int nb = 0; nb < 4; ++nb) o[nb] = zero8();

  const int qhi = q0 + 15;
  _Float16* pt = &lds[wave][0];

  for (int k0 = 0; k0 <= qhi; k0 += 64) {
    // prefetch next chunk's K rows and V region toward L2/WGP$
    if (k0 + 64 <= qhi) {
      __builtin_prefetch(kbase + (size_t)(k0 + 64 + lane) * DKn, 0, 3);
      __builtin_prefetch(vbase + (size_t)lane * Tn + k0 + 64, 0, 3);
      __builtin_prefetch(vbase + (size_t)(32 + lane) * Tn + k0 + 64, 0, 3);
    }

    float p_s[4][8];
    const bool diag = (k0 + 63 > q0);  // only diagonal chunks need causal cmp
#pragma unroll
    for (int st = 0; st < 4; ++st) {  // four 16-key subtiles
      v8f sc = zero8();
      const int key = k0 + st * 16 + n;
      const _Float16* krow = kbase + (size_t)key * DKn;
#pragma unroll
      for (int ec = 0; ec < 2; ++ec) {
        v8h blo = *(const v8h*)(krow + ec * 32 + half * 8);
        v8h bhi = *(const v8h*)(krow + ec * 32 + 16 + half * 8);
        v16h bf;
#pragma unroll
        for (int j = 0; j < 8; ++j) { bf[j] = blo[j]; bf[8 + j] = bhi[j]; }
        sc = wmma16x16x32(qa[ec], bf, sc);
      }
      float km = mbase[key];
      if (diag) {
#pragma unroll
        for (int v = 0; v < 8; ++v) {
          int qr = q0 + v + half * 8;
          float s = sc[v] * 0.125f;  // 1/sqrt(DK)
          if (key > qr || km == 0.0f) s = PADV;  // causal + key-pad mask
          p_s[st][v] = s;
        }
      } else {
#pragma unroll
        for (int v = 0; v < 8; ++v) {
          float s = sc[v] * 0.125f;
          if (km == 0.0f) s = PADV;  // key-pad mask only
          p_s[st][v] = s;
        }
      }
    }

    // online softmax across the 64 keys (rows live on 16 lanes per half)
    float sf[8];
#pragma unroll
    for (int v = 0; v < 8; ++v) {
      float rm = fmaxf(fmaxf(p_s[0][v], p_s[1][v]),
                       fmaxf(p_s[2][v], p_s[3][v]));
      rm = fmaxf(rm, __shfl_xor(rm, 1));
      rm = fmaxf(rm, __shfl_xor(rm, 2));
      rm = fmaxf(rm, __shfl_xor(rm, 4));
      rm = fmaxf(rm, __shfl_xor(rm, 8));
      float mnew = fmaxf(mi[v], rm);
      sf[v] = __expf(mi[v] - mnew);
      float rs = 0.0f;
#pragma unroll
      for (int st = 0; st < 4; ++st) {
        float p = __expf(p_s[st][v] - mnew);
        p_s[st][v] = p;
        rs += p;
      }
      rs += __shfl_xor(rs, 1);
      rs += __shfl_xor(rs, 2);
      rs += __shfl_xor(rs, 4);
      rs += __shfl_xor(rs, 8);
      li[v] = li[v] * sf[v] + rs;
      mi[v] = mnew;
    }
#pragma unroll
    for (int nb = 0; nb < 4; ++nb)
#pragma unroll
      for (int v = 0; v < 8; ++v) o[nb][v] *= sf[v];

    // C-layout -> A-layout transpose of P through LDS (wave-private tile)
#pragma unroll
    for (int st = 0; st < 4; ++st)
#pragma unroll
      for (int v = 0; v < 8; ++v)
        pt[(v + half * 8) * 72 + st * 16 + n] = (_Float16)p_s[st][v];

    // O += P * V over the 64 keys: two K=32 A-fragments from LDS
#pragma unroll
    for (int kc = 0; kc < 2; ++kc) {
      v16h pa;
      const _Float16* prow = pt + n * 72 + kc * 32;
      v8h plo = *(const v8h*)(prow + half * 8);
      v8h phi = *(const v8h*)(prow + 16 + half * 8);
#pragma unroll
      for (int j = 0; j < 8; ++j) {
        pa[j] = plo[j];
        pa[8 + j] = phi[j];
      }
#pragma unroll
      for (int nb = 0; nb < 4; ++nb) {
        const _Float16* vp =
            vbase + (size_t)(nb * 16 + n) * Tn + k0 + kc * 32 + half * 8;
        v8h vlo = *(const v8h*)vp, vhi = *(const v8h*)(vp + 16);
        v16h vf;
#pragma unroll
        for (int j = 0; j < 8; ++j) { vf[j] = vlo[j]; vf[8 + j] = vhi[j]; }
        o[nb] = wmma16x16x32(pa, vf, o[nb]);
      }
    }
  }

  // normalize and emit concatenated heads [B,T,H*DK] as f16
  _Float16* ob = heads + ((size_t)b * Tn + q0) * HD + h * DKn;
#pragma unroll
  for (int v = 0; v < 8; ++v) li[v] = 1.0f / li[v];
#pragma unroll
  for (int nb = 0; nb < 4; ++nb)
#pragma unroll
    for (int v = 0; v < 8; ++v)
      ob[(size_t)(v + half * 8) * HD + nb * 16 + n] =
          (_Float16)(o[nb][v] * li[v]);
}

// ---------------------------------------------------------------------------
// Output projection: heads[BT,1024] (f16) @ WoT[1024,1024] (f16, transposed)
// + bo -> out[BT,1024] f32. One wave: 16 x 64 tile.
// ---------------------------------------------------------------------------
__global__ __launch_bounds__(128) void out_kernel(
    const _Float16* __restrict__ heads, const _Float16* __restrict__ WoT,
    const float* __restrict__ bo, float* __restrict__ out) {
  const int lane = threadIdx.x & 31;
  const int wave = threadIdx.x >> 5;
  const int wid = blockIdx.x * 4 + wave;
  const int nct = Dn / 64;  // 16 column tiles
  const int r0 = (wid / nct) * 16;
  const int c0 = (wid % nct) * 64;
  const int n = lane & 15;
  const int half = lane >> 4;

  v8f acc[4];
#pragma unroll
  for (int nb = 0; nb < 4; ++nb) acc[nb] = zero8();

  const _Float16* arow = heads + (size_t)(r0 + n) * HD;
  for (int kk = 0; kk < HD; kk += 32) {
    v16h a;
    v8h lo = *(const v8h*)(arow + kk + half * 8);
    v8h hi = *(const v8h*)(arow + kk + 16 + half * 8);
#pragma unroll
    for (int j = 0; j < 8; ++j) {
      a[j] = lo[j];
      a[8 + j] = hi[j];
    }
#pragma unroll
    for (int nb = 0; nb < 4; ++nb) {
      const _Float16* bp =
          WoT + (size_t)(c0 + nb * 16 + n) * HD + kk + half * 8;
      v8h blo = *(const v8h*)bp, bhi = *(const v8h*)(bp + 16);
      v16h bf;
#pragma unroll
      for (int j = 0; j < 8; ++j) { bf[j] = blo[j]; bf[8 + j] = bhi[j]; }
      acc[nb] = wmma16x16x32(a, bf, acc[nb]);
    }
  }
#pragma unroll
  for (int nb = 0; nb < 4; ++nb) {
    float bias = bo[c0 + nb * 16 + n];
#pragma unroll
    for (int v = 0; v < 8; ++v)
      out[(size_t)(r0 + v + half * 8) * Dn + c0 + nb * 16 + n] =
          acc[nb][v] + bias;
  }
}

// ---------------------------------------------------------------------------
extern "C" void kernel_launch(void* const* d_in, const int* in_sizes, int n_in,
                              void* d_out, int out_size, void* d_ws,
                              size_t ws_size, hipStream_t stream) {
  (void)in_sizes; (void)n_in; (void)out_size; (void)ws_size;
  const float* q = (const float*)d_in[0];
  const float* k = (const float*)d_in[1];
  const float* v = (const float*)d_in[2];
  const float* Wq = (const float*)d_in[3];
  const float* bq = (const float*)d_in[4];
  const float* Wk = (const float*)d_in[5];
  const float* bk = (const float*)d_in[6];
  const float* Wv = (const float*)d_in[7];
  const float* bv = (const float*)d_in[8];
  const float* Wo = (const float*)d_in[9];
  const float* bo = (const float*)d_in[10];
  float* out = (float*)d_out;

  char* ws = (char*)d_ws;
  size_t off = 0;
  auto take = [&](size_t bytes) -> void* {
    off = (off + 255) & ~(size_t)255;
    void* p = ws + off;
    off += bytes;
    return p;
  };
  const size_t n_bhtd = (size_t)Bn * Hn * Tn * DKn;  // 16.8M elems
  _Float16* qs16 = (_Float16*)take(n_bhtd * 2);
  _Float16* ks16 = (_Float16*)take(n_bhtd * 2);
  _Float16* vsT16 = (_Float16*)take(n_bhtd * 2);
  _Float16* heads16 = (_Float16*)take(n_bhtd * 2);
  float* kmask = (float*)take((size_t)Bn * Hn * Tn * 4);
  _Float16* WqT = (_Float16*)take((size_t)Hn * Dn * DKn * 2);
  _Float16* WkT = (_Float16*)take((size_t)Hn * Dn * DKn * 2);
  _Float16* WvT = (_Float16*)take((size_t)Hn * Dn * DKn * 2);
  _Float16* WoT = (_Float16*)take((size_t)HD * Dn * 2);

  // 1) weight convert + transpose (K-dim contiguous for B-fragments)
  {
    size_t tw = (size_t)Hn * Dn * DKn;
    int blk = (int)((tw + 255) / 256);
    cvt_transpose_kernel<<<blk, 256, 0, stream>>>(Wq, WqT, Hn, Dn, DKn);
    cvt_transpose_kernel<<<blk, 256, 0, stream>>>(Wk, WkT, Hn, Dn, DKn);
    cvt_transpose_kernel<<<blk, 256, 0, stream>>>(Wv, WvT, Hn, Dn, DKn);
    size_t to = (size_t)HD * Dn;
    cvt_transpose_kernel<<<(int)((to + 255) / 256), 256, 0, stream>>>(
        Wo, WoT, 1, HD, Dn);
  }

  // 2) projections: B*H*(T/16) = 16384 waves, 4 waves/block
  const int proj_blocks = Bn * Hn * (Tn / 16) / 4;  // 4096
  proj_kernel<0><<<proj_blocks, 128, 0, stream>>>(q, WqT, bq, qs16, nullptr);
  proj_kernel<1><<<proj_blocks, 128, 0, stream>>>(k, WkT, bk, ks16, kmask);
  proj_kernel<2><<<proj_blocks, 128, 0, stream>>>(v, WvT, bv, vsT16, nullptr);

  // 3) fused flash attention
  attn_kernel<<<proj_blocks, 128, 0, stream>>>(qs16, ks16, vsT16, kmask,
                                               heads16);

  // 4) output projection
  const int out_blocks = (Bn * Tn / 16) * (Dn / 64) / 4;  // 4096
  out_kernel<<<out_blocks, 128, 0, stream>>>(heads16, WoT, bo, out);
}
